// EASSAAttention_39573828665394
// MI455X (gfx1250) — matrix operand order repaired and checked by reading, not compiled, not run on hardware
//
#include <hip/hip_runtime.h>
#include <hip/hip_bf16.h>
#include <math.h>

#define D_MODEL  512
#define N_HEADS  8
#define D_HEAD   64
#define M_STATES 64
#define S_LEN    4096
#define S_LOG2   12
#define LDP      65          // padded LDS row stride for scan state (gcd(65,64)=1)
#define NEGV     (-1e30f)

typedef __attribute__((ext_vector_type(16))) __bf16 v16bf;
typedef __attribute__((ext_vector_type(8)))  float  v8f;
typedef __attribute__((ext_vector_type(2)))  float  v2f;
typedef __attribute__((ext_vector_type(4)))  unsigned v4u;
typedef __attribute__((ext_vector_type(8)))  int      v8i;
typedef __attribute__((ext_vector_type(4)))  int      v4i;

#if __has_builtin(__builtin_amdgcn_tensor_load_to_lds)
#define EASSA_TDM 1
#endif

struct __attribute__((aligned(16))) F4 { float x, y, z, w; };
struct __attribute__((aligned(8)))  U2 { unsigned a, b; };
struct __attribute__((aligned(16))) H4 { unsigned a[4]; };
struct __attribute__((aligned(16))) H8 { H4 l, h; };

__device__ __forceinline__ unsigned pack2bf(float x, float y) {
    unsigned ux = __float_as_uint(x); ux += 0x7FFFu + ((ux >> 16) & 1u);
    unsigned uy = __float_as_uint(y); uy += 0x7FFFu + ((uy >> 16) & 1u);
    return (ux >> 16) | (uy & 0xFFFF0000u);
}
__device__ __forceinline__ __bf16 f2bf(float f) {
    unsigned u = __float_as_uint(f);
    u += 0x7FFFu + ((u >> 16) & 1u);
    union { unsigned short s; __bf16 b; } c; c.s = (unsigned short)(u >> 16);
    return c.b;
}

// ---------------------------------------------------------------------------
// One-time precision conversion kernels (amortized over all 4 GEMMs)
// ---------------------------------------------------------------------------
__global__ __launch_bounds__(256)
void convert_bf16(const float* __restrict__ in, __bf16* __restrict__ out) {
    size_t i = ((size_t)blockIdx.x * 256 + threadIdx.x) * 4;
    F4 f = *(const F4*)(in + i);
    U2 u; u.a = pack2bf(f.x, f.y); u.b = pack2bf(f.z, f.w);
    *(U2*)(out + i) = u;
}
// W[K,N] fp32 -> Wt[N,K] bf16 (transposed so B-tiles are row-major contiguous)
__global__ __launch_bounds__(256)
void transpose_to_bf16(const float* __restrict__ W, __bf16* __restrict__ Wt) {
    __shared__ float t[32][33];
    int bx = blockIdx.x * 32, by = blockIdx.y * 32;
    int lx = threadIdx.x & 31, ly = threadIdx.x >> 5;   // 8 rows per pass
    for (int i = 0; i < 32; i += 8)
        t[ly + i][lx] = W[(size_t)(by + ly + i) * D_MODEL + bx + lx];
    __syncthreads();
    for (int i = 0; i < 32; i += 8)
        Wt[(size_t)(bx + ly + i) * D_MODEL + by + lx] = f2bf(t[lx][ly + i]);
}

// ---------------------------------------------------------------------------
// TDM descriptor helpers (D# per CDNA5 ISA, groups 0/1; 2D tile, pad to
// 80B LDS row stride: pad_interval=3 (16 DW) pad_amount=3 (4 DW))
// ---------------------------------------------------------------------------
#ifdef EASSA_TDM
__device__ __forceinline__ v8i tdm_g1(int td0, int td1, int tile1) {
    v8i g;
    g[0] = (1 << 16) | (1 << 20) | (3 << 22) | (3 << 25);  // data_size=2B, pad
    g[1] = (td0 & 0xFFFF) << 16;                           // tensor_dim0 lo
    g[2] = ((unsigned)td0 >> 16) | ((td1 & 0xFFFF) << 16); // td0 hi | td1 lo
    g[3] = ((unsigned)td1 >> 16) | (32 << 16);             // td1 hi | tile_dim0=32
    g[4] = tile1;                                          // tile_dim1 (tile_dim2=0)
    g[5] = td0;                                            // dim0 stride lo (= row len)
    g[6] = 0; g[7] = 0;
    return g;
}
__device__ __forceinline__ void tdm_issue(const __bf16* gp, unsigned lds_off, v8i g1) {
    unsigned long long ga = (unsigned long long)(uintptr_t)gp;
    v4u g0;
    g0[0] = 1u;                                   // count=1 (valid user D#)
    g0[1] = lds_off;                              // LDS byte address
    g0[2] = (unsigned)ga;
    g0[3] = (unsigned)(ga >> 32) | (2u << 30);    // global addr hi | type=2
    v4i z4 = {0, 0, 0, 0};
    v8i z8 = {0, 0, 0, 0, 0, 0, 0, 0};
    // 6-arg (clang-23 / therock) form: g0, g1, g2, g3, <extra group>, cpol
    __builtin_amdgcn_tensor_load_to_lds(g0, g1, z4, z4, z8, 0);
}
#endif

// 32B fragment from two 16B-aligned LDS chunks: 2x ds_load_b128
__device__ __forceinline__ v16bf load_frag2(const __bf16* lo, const __bf16* hi) {
    H8 q; q.l = *(const H4*)lo; q.h = *(const H4*)hi;
    return __builtin_bit_cast(v16bf, q);
}

// ---------------------------------------------------------------------------
// GEMM: C[M,N] = A[M,K] @ Wt[N,K]^T + bias; bf16 in, fp32 out via WMMA.
// 256 threads = 8 waves; block tile 128(M) x 16(N); K-step 32.
// A/B tiles DMA'd by the Tensor Data Mover into double-buffered LDS (80B row
// stride via D# padding); per-operand fragment = 2x ds_load_b128.
// ---------------------------------------------------------------------------
template <int HS>
__global__ __launch_bounds__(256)
void gemm_bias_wmma(const __bf16* __restrict__ A, const __bf16* __restrict__ Wt,
                    const float* __restrict__ bias, float* __restrict__ C,
                    int Mrows, int N, int Kdim) {
    __shared__ __align__(16) __bf16 Asb[2 * 128 * 40];   // 2 x 10240B
    __shared__ __align__(16) __bf16 Bsb[2 * 16 * 40];    // 2 x 1280B

    const int tid  = threadIdx.x;
    const int wave = tid >> 5;
    const int lane = tid & 31;
    const int rowBase = blockIdx.x * 128;
    const int colBase = blockIdx.y * 16;

    v8f acc = {0.f, 0.f, 0.f, 0.f, 0.f, 0.f, 0.f, 0.f};

    const int mr = wave * 16 + (lane & 15);
    const int nn = lane & 15;
    const int hh = lane >> 4;
    const int nK = Kdim / 32;

#ifdef EASSA_TDM
    const v8i g1A = tdm_g1(Kdim, Mrows, 128);
    const v8i g1B = tdm_g1(Kdim, N, 16);
    const unsigned ldsA[2] = {(unsigned)(uintptr_t)&Asb[0], (unsigned)(uintptr_t)&Asb[128 * 40]};
    const unsigned ldsB[2] = {(unsigned)(uintptr_t)&Bsb[0], (unsigned)(uintptr_t)&Bsb[16 * 40]};
    const __bf16* Agp = A  + (size_t)rowBase * Kdim;
    const __bf16* Bgp = Wt + (size_t)colBase * Kdim;

    if (tid < 32) {                       // one TDM issue per tile, wave 0 only
        tdm_issue(Agp, ldsA[0], g1A);
        tdm_issue(Bgp, ldsB[0], g1B);
    }
    for (int it = 0; it < nK; ++it) {
        const int bi = it & 1;
        if (tid < 32) __builtin_amdgcn_s_wait_tensorcnt(0);   // tile it landed
        __syncthreads();                  // all waves: done reading buf bi^1
        if (tid < 32 && it + 1 < nK) {    // prefetch tile it+1 during compute
            tdm_issue(Agp + (it + 1) * 32, ldsA[bi ^ 1], g1A);
            tdm_issue(Bgp + (it + 1) * 32, ldsB[bi ^ 1], g1B);
        }
        const __bf16* Ab = &Asb[bi * 128 * 40];
        const __bf16* Bb = &Bsb[bi * 16 * 40];
        v16bf a = load_frag2(&Ab[mr * 40 + 8 * hh], &Ab[mr * 40 + 16 + 8 * hh]);
        v16bf b = load_frag2(&Bb[nn * 40 + 8 * hh], &Bb[nn * 40 + 16 + 8 * hh]);
        acc = __builtin_amdgcn_wmma_f32_16x16x32_bf16(
                  false, a, false, b, (short)0, acc, false, false);
    }
#else
    for (int it = 0; it < nK; ++it) {
        const int k0 = it * 32;
        // A: 512 chunks of 8 halfs (16B), 2 per thread
        #pragma unroll
        for (int i = 0; i < 2; ++i) {
            int c = i * 256 + tid;
            int r = c >> 2, part = c & 3;
            *(H4*)&Asb[r * 40 + part * 8] =
                *(const H4*)(A + (size_t)(rowBase + r) * Kdim + k0 + part * 8);
        }
        if (tid < 64) {                    // B: 64 chunks
            int r = tid >> 2, part = tid & 3;
            *(H4*)&Bsb[r * 40 + part * 8] =
                *(const H4*)(Wt + (size_t)(colBase + r) * Kdim + k0 + part * 8);
        }
        __syncthreads();
        v16bf a = load_frag2(&Asb[mr * 40 + 8 * hh], &Asb[mr * 40 + 16 + 8 * hh]);
        v16bf b = load_frag2(&Bsb[nn * 40 + 8 * hh], &Bsb[nn * 40 + 16 + 8 * hh]);
        acc = __builtin_amdgcn_wmma_f32_16x16x32_bf16(
                  false, a, false, b, (short)0, acc, false, false);
        __syncthreads();
    }
#endif

    const int col = colBase + nn;
    const float bv = bias[col];
    #pragma unroll
    for (int i = 0; i < 8; ++i) {
        int row = rowBase + wave * 16 + i + 8 * hh;
        float v = acc[i] + bv;
        size_t idx;
        if (HS) {    // compile-time shifts: S=4096, dh=64
            int bb = row >> S_LOG2, ss = row & (S_LEN - 1);
            int hd = col >> 6,      d  = col & 63;
            idx = (((size_t)(bb * N_HEADS + hd) * S_LEN + ss) * D_HEAD) + d;
        } else {
            idx = (size_t)row * N + col;
        }
        C[idx] = v;
    }
}

// ---------------------------------------------------------------------------
// Block-wide (64-thread / 2-wave32) reductions: shfl_xor + 2-slot LDS combine
// ---------------------------------------------------------------------------
__device__ __forceinline__ float blockSum(float v, float* fs, int tid) {
    #pragma unroll
    for (int m = 16; m >= 1; m >>= 1) v += __shfl_xor(v, m, 32);
    if ((tid & 31) == 0) fs[tid >> 5] = v;
    __syncthreads();
    float r = fs[0] + fs[1];
    __syncthreads();
    return r;
}
__device__ __forceinline__ float blockMax(float v, float* fs, int tid) {
    #pragma unroll
    for (int m = 16; m >= 1; m >>= 1) v = fmaxf(v, __shfl_xor(v, m, 32));
    if ((tid & 31) == 0) fs[tid >> 5] = v;
    __syncthreads();
    float r = fmaxf(fs[0], fs[1]);
    __syncthreads();
    return r;
}
__device__ __forceinline__ void blockArgMax(float& v, int& idx, float* fs, int* is, int tid) {
    #pragma unroll
    for (int m = 16; m >= 1; m >>= 1) {
        float ov = __shfl_xor(v, m, 32);
        int   oi = __shfl_xor(idx, m, 32);
        if (ov > v || (ov == v && oi < idx)) { v = ov; idx = oi; }
    }
    if ((tid & 31) == 0) { fs[tid >> 5] = v; is[tid >> 5] = idx; }
    __syncthreads();
    float v0 = fs[0], v1 = fs[1]; int i0 = is[0], i1 = is[1];
    if (v1 > v0 || (v1 == v0 && i1 < i0)) { v = v1; idx = i1; }
    else                                  { v = v0; idx = i0; }
    __syncthreads();
}
__device__ __forceinline__ int blockMinI(int v, int* is, int tid) {
    #pragma unroll
    for (int m = 16; m >= 1; m >>= 1) v = min(v, __shfl_xor(v, m, 32));
    if ((tid & 31) == 0) is[tid >> 5] = v;
    __syncthreads();
    int r = min(is[0], is[1]);
    __syncthreads();
    return r;
}

// ---------------------------------------------------------------------------
// Sequential EASSA scan. One block of 64 threads (2 wave32) per (b,h) chain.
// State resident in LDS; merge pairwise-sim matrix via V_WMMA_F32_16X16X4_F32
// under the workgroup-uniform need_merge branch (EXEC all-1s there).
// ---------------------------------------------------------------------------
__global__ __launch_bounds__(64)
void eassa_scan(const float* __restrict__ Q, const float* __restrict__ K,
                const float* __restrict__ V, const float* __restrict__ ec_w,
                const float* __restrict__ ec_b, float* __restrict__ O) {
    __shared__ float cent[M_STATES * LDP];
    __shared__ float st  [M_STATES * LDP];
    __shared__ float cn  [M_STATES * LDP];
    __shared__ float pm  [M_STATES * LDP];
    __shared__ float cnt [M_STATES];
    __shared__ float kt[D_HEAD], qt[D_HEAD], vt[D_HEAD], prob[M_STATES];
    __shared__ float pf[2];
    __shared__ int   pi[2];

    const int tid = threadIdx.x;            // 0..63: state id == dim id
    const int bh  = blockIdx.x;
    const int h   = bh & (N_HEADS - 1);
    const float base_budget = 100.0f / (float)S_LEN;
    const float inv_scale   = 0.125f;       // 1/sqrt(64)

    for (int i = tid; i < M_STATES * LDP; i += 64) { cent[i] = 0.f; st[i] = 0.f; }
    cnt[tid] = 0.f;
    __syncthreads();

    const float ecw_t = ec_w[h * D_HEAD + tid];
    const float ecb_h = ec_b[h];
    const float* Qp = Q + (size_t)bh * S_LEN * D_HEAD;
    const float* Kp = K + (size_t)bh * S_LEN * D_HEAD;
    const float* Vp = V + (size_t)bh * S_LEN * D_HEAD;
    float*       Op = O + (size_t)bh * S_LEN * D_HEAD;

    for (int s = 0; s < S_LEN; ++s) {
        const int off = s * D_HEAD + tid;
        float kv = Kp[off], qv = Qp[off], vv = Vp[off];
        kt[tid] = kv; qt[tid] = qv; vt[tid] = vv;
        if (s + 1 < S_LEN) {                 // hide next step's 768B behind this step
            __builtin_prefetch(Kp + off + D_HEAD, 0, 1);
            __builtin_prefetch(Qp + off + D_HEAD, 0, 1);
            __builtin_prefetch(Vp + off + D_HEAD, 0, 1);
        }

        // budget = base*2*sigmoid(k.ec_w + ec_b); 1/(||k||+eps)
        float zdot = blockSum(kv * ecw_t, pf, tid);   // barrier also publishes kt/qt/vt
        float ksq  = blockSum(kv * kv,   pf, tid);
        float bud  = base_budget * 2.0f / (1.0f + __expf(-(zdot + ecb_h)));
        float kinv = 1.0f / (sqrtf(ksq) + 1e-12f);

        // normalize own centroid row, cosine sim with normalized key
        float dot = 0.f;
        {
            float ss = 0.f;
            for (int d = 0; d < D_HEAD; ++d) { float c = cent[tid * LDP + d]; ss += c * c; }
            float cinv = 1.0f / (sqrtf(ss) + 1e-12f);
            for (int d = 0; d < D_HEAD; ++d) {
                float c = cent[tid * LDP + d] * cinv;
                cn[tid * LDP + d] = c;
                dot += c * kt[d];
            }
        }
        bool valid = cnt[tid] > 0.f;
        float Kn = blockSum(valid ? 1.f : 0.f, pf, tid);

        float best = valid ? dot * kinv : NEGV;
        int best_idx = tid;
        blockArgMax(best, best_idx, pf, pi, tid);

        // all threads compute routing decision redundantly (uniform)
        const float lam = 1.0f / fmaxf(bud, 1e-6f);
        const int create     = (Kn == 0.0f) || ((best - lam) > best);
        const int need_merge = create && (Kn >= (float)M_STATES);

        if (need_merge) {   // uniform branch -> EXEC all-1s (WMMA legal)
            const int wave = tid >> 5, lane = tid & 31;
            const int mrow = lane & 15, hq = lane >> 4;
            for (int t = 0; t < 8; ++t) {
                const int tile = wave * 8 + t;
                const int rb = (tile >> 2) * 16, cb = (tile & 3) * 16;
                v8f pacc = {0.f, 0.f, 0.f, 0.f, 0.f, 0.f, 0.f, 0.f};
                #pragma unroll
                for (int k4 = 0; k4 < D_HEAD; k4 += 4) {
                    const int kk = k4 + 2 * hq;
                    v2f fa, fb;
                    fa[0] = cn[(rb + mrow) * LDP + kk];
                    fa[1] = cn[(rb + mrow) * LDP + kk + 1];
                    fb[0] = cn[(cb + mrow) * LDP + kk];   // cn @ cn^T
                    fb[1] = cn[(cb + mrow) * LDP + kk + 1];
                    pacc = __builtin_amdgcn_wmma_f32_16x16x4_f32(
                               false, fa, false, fb, (short)0, pacc, false, false);
                }
                #pragma unroll
                for (int i = 0; i < 8; ++i)
                    pm[(rb + i + 8 * hq) * LDP + cb + mrow] = pacc[i];
            }
            __syncthreads();
            // masked flat argmax over pm (lowest flat index wins ties)
            float bv = NEGV; int bc = 0;
            for (int c = 0; c < M_STATES; ++c) {
                bool ok = valid && (cnt[c] > 0.f) && (c != tid);
                float pv = ok ? pm[tid * LDP + c] : NEGV;
                if (pv > bv) { bv = pv; bc = c; }
            }
            int flat = tid * M_STATES + bc;
            blockArgMax(bv, flat, pf, pi, tid);
            const int ra = flat >> 6, rb2 = flat & 63;
            const int mi = min(ra, rb2), mj = max(ra, rb2);
            const float ni = cnt[mi], nj = cnt[mj];
            const float nt = ni + nj;
            const float dn = fmaxf(nt, 1.0f);
            const float mc = (ni * cent[mi * LDP + tid] + nj * cent[mj * LDP + tid]) / dn;
            const float ms = (ni * st[mi * LDP + tid]   + nj * st[mj * LDP + tid])   / dn;
            __syncthreads();   // all reads done before writes
            cent[mi * LDP + tid] = mc;  cent[mj * LDP + tid] = 0.f;
            st[mi * LDP + tid]   = ms;  st[mj * LDP + tid]   = 0.f;
            if (tid == 0) { cnt[mi] = nt; cnt[mj] = 0.f; }
            __syncthreads();
        }

        if (create) {
            int fv = (cnt[tid] > 0.f) ? M_STATES : tid;   // first free slot
            int fr = blockMinI(fv, pi, tid);
            const int f = (fr >= M_STATES) ? 0 : fr;
            cent[f * LDP + tid] = kt[tid];
            st[f * LDP + tid]   = vt[tid];
            if (tid == 0) cnt[f] = 1.0f;
        } else {
            const float n = cnt[best_idx];
            __syncthreads();   // all read n before thread 0 overwrites
            const float inv = 1.0f / (n + 1.0f);
            cent[best_idx * LDP + tid] += (kt[tid] - cent[best_idx * LDP + tid]) * inv;
            st[best_idx * LDP + tid]   += (vt[tid] - st[best_idx * LDP + tid])   * inv;
            if (tid == 0) cnt[best_idx] = n + 1.0f;
        }
        __syncthreads();

        // online attention of q over current states
        float qdot = 0.f;
        for (int d = 0; d < D_HEAD; ++d) qdot += cent[tid * LDP + d] * qt[d];
        float lg = (cnt[tid] > 0.f) ? qdot * inv_scale : NEGV;
        float mx = blockMax(lg, pf, tid);
        float e  = __expf(lg - mx);
        prob[tid] = e;
        float denom = blockSum(e, pf, tid);   // barrier publishes prob[]
        {
            float oacc = 0.f;
            for (int m = 0; m < M_STATES; ++m) oacc += prob[m] * st[m * LDP + tid];
            Op[off] = oacc / denom;
        }
        __syncthreads();
    }
}

// ---------------------------------------------------------------------------
extern "C" void kernel_launch(void* const* d_in, const int* in_sizes, int n_in,
                              void* d_out, int out_size, void* d_ws, size_t ws_size,
                              hipStream_t stream) {
    (void)n_in; (void)out_size; (void)ws_size;
    const float* x    = (const float*)d_in[0];
    const float* wq   = (const float*)d_in[1];
    const float* bq   = (const float*)d_in[2];
    const float* wk   = (const float*)d_in[3];
    const float* bk   = (const float*)d_in[4];
    const float* wv   = (const float*)d_in[5];
    const float* bv   = (const float*)d_in[6];
    const float* wo   = (const float*)d_in[7];
    const float* bo   = (const float*)d_in[8];
    const float* ec_w = (const float*)d_in[9];
    const float* ec_b = (const float*)d_in[10];

    const int D = D_MODEL;
    const int Mrows = in_sizes[0] / D;        // B*S
    const int B = Mrows / S_LEN;
    const size_t MD = (size_t)Mrows * D;

    float*  qbuf = (float*)d_ws;
    float*  kbuf = qbuf + MD;
    float*  vbuf = kbuf + MD;
    float*  obuf = vbuf + MD;
    __bf16* xb   = (__bf16*)(obuf + MD);
    __bf16* ob   = xb + MD;
    __bf16* wtq  = ob + MD;
    __bf16* wtk  = wtq + (size_t)D * D;
    __bf16* wtv  = wtk + (size_t)D * D;
    __bf16* wto  = wtv + (size_t)D * D;

    // one-time bf16 conversions (x) and transposed-bf16 weights
    convert_bf16<<<(unsigned)(MD / 1024), 256, 0, stream>>>(x, xb);
    dim3 tg(D / 32, D / 32);
    transpose_to_bf16<<<tg, 256, 0, stream>>>(wq, wtq);
    transpose_to_bf16<<<tg, 256, 0, stream>>>(wk, wtk);
    transpose_to_bf16<<<tg, 256, 0, stream>>>(wv, wtv);
    transpose_to_bf16<<<tg, 256, 0, stream>>>(wo, wto);

    dim3 grid(Mrows / 128, D / 16), blk(256);
    gemm_bias_wmma<1><<<grid, blk, 0, stream>>>(xb, wtq, bq, qbuf, Mrows, D, D);
    gemm_bias_wmma<1><<<grid, blk, 0, stream>>>(xb, wtk, bk, kbuf, Mrows, D, D);
    gemm_bias_wmma<1><<<grid, blk, 0, stream>>>(xb, wtv, bv, vbuf, Mrows, D, D);

    eassa_scan<<<B * N_HEADS, 64, 0, stream>>>(qbuf, kbuf, vbuf, ec_w, ec_b, obuf);

    convert_bf16<<<(unsigned)(MD / 1024), 256, 0, stream>>>(obuf, ob);
    gemm_bias_wmma<0><<<grid, blk, 0, stream>>>(ob, wto, bo, (float*)d_out, Mrows, D, D);
}
